// Model_77223511982479
// MI455X (gfx1250) — 8-bit matrix operand order repaired and checked by reading, compiled, not run on hardware
//
#include <hip/hip_runtime.h>
#include <hip/hip_bf16.h>

typedef __attribute__((ext_vector_type(8))) int v8i;

#define OFF_W1Q   256ull
#define OFF_W2Q   512ull
#define OFF_FW1Q  4096ull
#define OFF_FW2Q  36864ull
#define OFF_FW3Q  49152ull
#define OFF_QX    65536ull
#define OFF_ACC1  6488064ull
#define OFF_Q1    20643840ull
#define OFF_ACC2  65536ull
#define OFF_Q2    16842752ull
#define OFF_ACC3  20643840ull
#define OFF_ACC4  65536ull
#define OFF_ACC5  24838144ull

__device__ __forceinline__ int imax(int a, int b) { return a > b ? a : b; }

__device__ __forceinline__ int K0A(int j, int hi) {
  return ((j >> 2) << 5) | (((j >> 1) & 1) << 4) | ((j & 1) << 2) | (hi << 3);
}
__device__ __forceinline__ int K0B(int j, int hi) {
  return ((j >> 2) << 5) | (((j >> 1) & 1) << 4) | ((j & 1) << 2) | (hi << 3);
}

__device__ __forceinline__ float s1_of(const char* ws) {
  return __uint_as_float(((const unsigned*)ws)[0]) * (1.0f / 7.0f);
}

__global__ void k_zero(char* ws) {
  if (threadIdx.x < 16) ((int*)ws)[threadIdx.x] = 0;
}

__global__ __launch_bounds__(256) void k_absmax(const float* __restrict__ x,
                                                char* __restrict__ ws, int n) {
  __shared__ float red[256];
  float m = 0.0f;
  for (int i = blockIdx.x * 256 + threadIdx.x; i < n; i += gridDim.x * 256)
    m = fmaxf(m, fabsf(x[i]));
  int t = threadIdx.x;
  red[t] = m; __syncthreads();
  for (int s = 128; s > 0; s >>= 1) {
    if (t < s) red[t] = fmaxf(red[t], red[t + s]);
    __syncthreads();
  }
  if (t == 0) atomicMax((unsigned*)ws, __float_as_uint(red[0]));
}

__device__ __forceinline__ signed char q3clip(float v, float inv) {
  float r = rintf(v * inv);
  return (signed char)fminf(3.0f, fmaxf(-3.0f, r));
}

__global__ __launch_bounds__(256) void k_wquant(const float* __restrict__ w1,
                                                const float* __restrict__ w2,
                                                const float* __restrict__ fw1,
                                                const float* __restrict__ fw2,
                                                const float* __restrict__ fw3,
                                                char* __restrict__ ws) {
  __shared__ float red[256];
  __shared__ float sinv;
  int b = blockIdx.x, t = threadIdx.x;
  const float* src; int n;
  if      (b == 0) { src = w1;  n = 150;   }
  else if (b == 1) { src = w2;  n = 2400;  }
  else if (b == 2) { src = fw1; n = 30720; }
  else if (b == 3) { src = fw2; n = 10080; }
  else             { src = fw3; n = 840;   }
  float m = 0.0f;
  for (int i = t; i < n; i += 256) m = fmaxf(m, fabsf(src[i]));
  red[t] = m; __syncthreads();
  for (int s = 128; s > 0; s >>= 1) {
    if (t < s) red[t] = fmaxf(red[t], red[t + s]);
    __syncthreads();
  }
  if (t == 0) {
    float sc = red[0] * (1.0f / 3.0f);
    ((float*)ws)[6 + b] = sc;
    sinv = 1.0f / sc;
  }
  __syncthreads();
  float inv = sinv;
  if (b == 0) {
    signed char* dst = (signed char*)(ws + OFF_W1Q);
    for (int i = t; i < 160; i += 256)
      dst[i] = (i < 150) ? q3clip(src[i], inv) : (signed char)0;
  } else if (b == 1) {
    signed char* dst = (signed char*)(ws + OFF_W2Q);
    for (int i = t; i < 16 * 192; i += 256) {
      int oc = i / 192, k = i - oc * 192;
      dst[i] = (k < 150) ? q3clip(src[oc * 150 + k], inv) : (signed char)0;
    }
  } else if (b == 2) {
    signed char* dst = (signed char*)(ws + OFF_FW1Q);
    for (int i = t; i < 128 * 256; i += 256) {
      int o = i >> 8, k = i & 255;
      dst[i] = (o < 120) ? q3clip(src[o * 256 + k], inv) : (signed char)0;
    }
  } else if (b == 3) {
    signed char* dst = (signed char*)(ws + OFF_FW2Q);
    for (int i = t; i < 96 * 128; i += 256) {
      int o = i >> 7, k = i & 127;
      dst[i] = (o < 84 && k < 120) ? q3clip(src[o * 120 + k], inv) : (signed char)0;
    }
  } else {
    signed char* dst = (signed char*)(ws + OFF_FW3Q);
    for (int i = t; i < 16 * 128; i += 256) {
      int o = i >> 7, k = i & 127;
      dst[i] = (o < 10 && k < 84) ? q3clip(src[o * 84 + k], inv) : (signed char)0;
    }
  }
}

__global__ __launch_bounds__(256) void k_qx(const float* __restrict__ x,
                                            char* __restrict__ ws, int n) {
  int i = blockIdx.x * 256 + threadIdx.x;
  if (i >= n) return;
  float inv = 1.0f / s1_of(ws);
  float q = rintf(x[i] * inv);
  q = fminf(7.0f, fmaxf(-8.0f, q));
  ((signed char*)(ws + OFF_QX))[i] = (signed char)q;
}

__global__ __launch_bounds__(256) void k_conv1(const float* __restrict__ b1,
                                               char* __restrict__ ws) {
  int t = blockIdx.x * 256 + threadIdx.x;
  int px = t % 12, u = t / 12;
  int py = u % 12; u /= 12;
  int oc = u % 6;  int b = u / 6;
  float s1 = s1_of(ws);
  float sw1 = ((const float*)ws)[6];
  int b1i = (int)rintf(b1[oc] / (s1 * sw1));
  const signed char* qx = (const signed char*)(ws + OFF_QX) + (size_t)b * 784;
  const signed char* wq = (const signed char*)(ws + OFF_W1Q) + oc * 25;
  int wr[25];
#pragma unroll
  for (int k = 0; k < 25; k++) wr[k] = wq[k];
  int best = -2147483647;
#pragma unroll
  for (int dy = 0; dy < 2; dy++)
#pragma unroll
    for (int dx = 0; dx < 2; dx++) {
      int oy = 2 * py + dy, ox = 2 * px + dx;
      int acc = b1i;
#pragma unroll
      for (int ky = 0; ky < 5; ky++)
#pragma unroll
        for (int kx = 0; kx < 5; kx++)
          acc += (int)qx[(oy + ky) * 28 + (ox + kx)] * wr[ky * 5 + kx];
      best = imax(best, acc);
    }
  ((short*)(ws + OFF_ACC1))[t] = (short)best;
  __shared__ int red[256];
  int tid = threadIdx.x;
  red[tid] = imax(best, 0); __syncthreads();
  for (int s = 128; s > 0; s >>= 1) {
    if (tid < s) red[tid] = imax(red[tid], red[tid + s]);
    __syncthreads();
  }
  if (tid == 0) atomicMax(((int*)ws) + 1, red[0]);
}

__global__ __launch_bounds__(256) void k_quant1(char* __restrict__ ws) {
  int t = blockIdx.x * 256 + threadIdx.x;
  int v = ((const short*)(ws + OFF_ACC1))[t];
  float inv = 15.0f / (float)imax(((const int*)ws)[1], 1);
  float q = rintf((float)v * inv);
  q = fminf(15.0f, fmaxf(0.0f, q));
  ((signed char*)(ws + OFF_Q1))[t] = (signed char)q;
}

__global__ __launch_bounds__(256) void k_conv2(const float* __restrict__ b2,
                                               char* __restrict__ ws) {
  __shared__ int raw_i[432];
  __shared__ int im2_i[6144];
  __shared__ int wl_i[768];
  __shared__ int red[256];
  signed char* raw = (signed char*)raw_i;
  signed char* im2 = (signed char*)im2_i;
  signed char* wl  = (signed char*)wl_i;
  int tid = threadIdx.x;
  int img0 = blockIdx.x * 2;
  const int* q1 = (const int*)(ws + OFF_Q1 + (size_t)img0 * 864);
  for (int i = tid; i < 432; i += 256) raw_i[i] = q1[i];
  const int* wsrc = (const int*)(ws + OFF_W2Q);
  for (int i = tid; i < 768; i += 256) wl_i[i] = wsrc[i];
  __syncthreads();
  for (int i = tid; i < 24576; i += 256) {
    int k = i % 192;
    int pos = (i / 192) & 63;
    int img = i / 12288;
    signed char v = 0;
    if (k < 150) {
      int c = k / 25, r = k - c * 25, ky = r / 5, kx = r - ky * 5;
      v = raw[img * 864 + c * 144 + ((pos >> 3) + ky) * 12 + (pos & 7) + kx];
    }
    im2[i] = v;
  }
  __syncthreads();
  int lane = tid & 31, wave = tid >> 5;
  int hi = lane >> 4, n = lane & 15;
  int img = wave >> 2, posBase = (wave & 3) * 16;
  const int* si = (const int*)ws; const float* sf = (const float*)ws;
  float s2 = s1_of(ws) * sf[6] * (float)imax(si[1], 1) * (1.0f / 15.0f);
  int b2i = (int)rintf(b2[n] / (s2 * sf[7]));
  v8i c;
#pragma unroll
  for (int j = 0; j < 8; j++) c[j] = b2i;
  const signed char* arow = im2 + (img * 64 + posBase + n) * 192;
  const signed char* brow = wl + n * 192;
#pragma unroll
  for (int kc = 0; kc < 3; kc++) {
    v8i a, bm;
#pragma unroll
    for (int j = 0; j < 8; j++) {
      a[j]  = *(const int*)(arow + kc * 64 + K0A(j, hi));
      bm[j] = *(const int*)(brow + kc * 64 + K0B(j, hi));
    }
    c = __builtin_amdgcn_wmma_i32_16x16x64_iu8(false, a, true, bm, c, false, false);
  }
  short* acc2 = (short*)(ws + OFF_ACC2);
  size_t obase = (((size_t)(img0 + img)) * 16 + n) * 64 + posBase + (hi << 3);
  int m = 0;
#pragma unroll
  for (int j = 0; j < 8; j++) {
    int v = c[j];
    acc2[obase + j] = (short)v;
    m = imax(m, v);
  }
  red[tid] = imax(m, 0); __syncthreads();
  for (int s = 128; s > 0; s >>= 1) {
    if (tid < s) red[tid] = imax(red[tid], red[tid + s]);
    __syncthreads();
  }
  if (tid == 0) atomicMax(((int*)ws) + 2, red[0]);
}

__global__ __launch_bounds__(256) void k_poolq2(char* __restrict__ ws) {
  int t = blockIdx.x * 256 + threadIdx.x;
  int px = t & 3, py = (t >> 2) & 3, cch = (t >> 4) & 15, b = t >> 8;
  const short* acc2 = (const short*)(ws + OFF_ACC2);
  size_t base = ((size_t)b * 16 + cch) * 64 + (py * 2) * 8 + px * 2;
  int v = acc2[base];
  v = imax(v, (int)acc2[base + 1]);
  v = imax(v, (int)acc2[base + 8]);
  v = imax(v, (int)acc2[base + 9]);
  float inv = 15.0f / (float)imax(((const int*)ws)[2], 1);
  float q = rintf((float)v * inv);
  q = fminf(15.0f, fmaxf(0.0f, q));
  ((signed char*)(ws + OFF_Q2))[t] = (signed char)q;
}

__global__ __launch_bounds__(256) void k_fc1(const float* __restrict__ fb1,
                                             char* __restrict__ ws) {
  __shared__ int a_i[1024];
  __shared__ int red[256];
  signed char* a_lds = (signed char*)a_i;
  int tid = threadIdx.x;
  int m0 = blockIdx.x * 16;
  const int* q2 = (const int*)(ws + OFF_Q2 + (size_t)m0 * 256);
  for (int i = tid; i < 1024; i += 256) a_i[i] = q2[i];
  __syncthreads();
  int lane = tid & 31, wave = tid >> 5, hi = lane >> 4, nn = lane & 15;
  int n = wave * 16 + nn;
  const int* si = (const int*)ws; const float* sf = (const float*)ws;
  float s2 = s1_of(ws) * sf[6] * (float)imax(si[1], 1) / 15.0f;
  float s3 = s2 * sf[7] * (float)imax(si[2], 1) / 15.0f;
  int bi = (n < 120) ? (int)rintf(fb1[n] / (s3 * sf[8])) : 0;
  v8i c;
#pragma unroll
  for (int j = 0; j < 8; j++) c[j] = bi;
  const signed char* arow = a_lds + nn * 256;
  const signed char* brow = (const signed char*)(ws + OFF_FW1Q) + n * 256;
#pragma unroll
  for (int kc = 0; kc < 4; kc++) {
    v8i a, bm;
#pragma unroll
    for (int j = 0; j < 8; j++) {
      a[j]  = *(const int*)(arow + kc * 64 + K0A(j, hi));
      bm[j] = *(const int*)(brow + kc * 64 + K0B(j, hi));
    }
    c = __builtin_amdgcn_wmma_i32_16x16x64_iu8(false, a, true, bm, c, false, false);
  }
  int* acc3 = (int*)(ws + OFF_ACC3);
  int m = 0;
#pragma unroll
  for (int j = 0; j < 8; j++) {
    int M = j + (hi << 3);
    acc3[(size_t)(m0 + M) * 128 + n] = c[j];
    m = imax(m, c[j]);
  }
  red[tid] = imax(m, 0); __syncthreads();
  for (int s = 128; s > 0; s >>= 1) {
    if (tid < s) red[tid] = imax(red[tid], red[tid + s]);
    __syncthreads();
  }
  if (tid == 0) atomicMax(((int*)ws) + 3, red[0]);
}

__global__ __launch_bounds__(192) void k_fc2(const float* __restrict__ fb2,
                                             char* __restrict__ ws) {
  __shared__ int a_i[512];
  __shared__ int red[256];
  signed char* a_lds = (signed char*)a_i;
  int tid = threadIdx.x;
  int m0 = blockIdx.x * 16;
  const int* si = (const int*)ws; const float* sf = (const float*)ws;
  float inv3 = 15.0f / (float)imax(si[3], 1);
  const int* acc3 = (const int*)(ws + OFF_ACC3);
  for (int i = tid; i < 2048; i += 192) {
    int m = i >> 7, k = i & 127;
    float q = rintf((float)acc3[(size_t)(m0 + m) * 128 + k] * inv3);
    a_lds[i] = (signed char)fminf(15.0f, fmaxf(0.0f, q));
  }
  __syncthreads();
  int lane = tid & 31, wave = tid >> 5, hi = lane >> 4, nn = lane & 15;
  int n = wave * 16 + nn;
  float s2 = s1_of(ws) * sf[6] * (float)imax(si[1], 1) / 15.0f;
  float s3 = s2 * sf[7] * (float)imax(si[2], 1) / 15.0f;
  float s4 = s3 * sf[8] * (float)imax(si[3], 1) / 15.0f;
  int bi = (n < 84) ? (int)rintf(fb2[n] / (s4 * sf[9])) : 0;
  v8i c;
#pragma unroll
  for (int j = 0; j < 8; j++) c[j] = bi;
  const signed char* arow = a_lds + nn * 128;
  const signed char* brow = (const signed char*)(ws + OFF_FW2Q) + n * 128;
#pragma unroll
  for (int kc = 0; kc < 2; kc++) {
    v8i a, bm;
#pragma unroll
    for (int j = 0; j < 8; j++) {
      a[j]  = *(const int*)(arow + kc * 64 + K0A(j, hi));
      bm[j] = *(const int*)(brow + kc * 64 + K0B(j, hi));
    }
    c = __builtin_amdgcn_wmma_i32_16x16x64_iu8(false, a, true, bm, c, false, false);
  }
  int* acc4 = (int*)(ws + OFF_ACC4);
  int m = 0;
#pragma unroll
  for (int j = 0; j < 8; j++) {
    int M = j + (hi << 3);
    acc4[(size_t)(m0 + M) * 96 + n] = c[j];
    m = imax(m, c[j]);
  }
  red[tid] = imax(m, 0);
  if (tid < 64) red[192 + tid] = 0;
  __syncthreads();
  for (int s = 128; s > 0; s >>= 1) {
    if (tid < s) red[tid] = imax(red[tid], red[tid + s]);
    __syncthreads();
  }
  if (tid == 0) atomicMax(((int*)ws) + 4, red[0]);
}

__global__ __launch_bounds__(256) void k_fc3(char* __restrict__ ws) {
  __shared__ int a_i[4096];
  __shared__ int red[256];
  signed char* a_lds = (signed char*)a_i;
  int tid = threadIdx.x;
  int m0 = blockIdx.x * 128;
  const int* si = (const int*)ws;
  float inv4 = 15.0f / (float)imax(si[4], 1);
  const int* acc4 = (const int*)(ws + OFF_ACC4);
  for (int i = tid; i < 16384; i += 256) {
    int m = i >> 7, k = i & 127;
    signed char q = 0;
    if (k < 96) {
      float t = rintf((float)acc4[(size_t)(m0 + m) * 96 + k] * inv4);
      q = (signed char)fminf(15.0f, fmaxf(0.0f, t));
    }
    a_lds[i] = q;
  }
  __syncthreads();
  int lane = tid & 31, wave = tid >> 5, hi = lane >> 4, nn = lane & 15;
  v8i c;
#pragma unroll
  for (int j = 0; j < 8; j++) c[j] = 0;
  const signed char* arow = a_lds + (wave * 16 + nn) * 128;
  const signed char* brow = (const signed char*)(ws + OFF_FW3Q) + nn * 128;
#pragma unroll
  for (int kc = 0; kc < 2; kc++) {
    v8i a, bm;
#pragma unroll
    for (int j = 0; j < 8; j++) {
      a[j]  = *(const int*)(arow + kc * 64 + K0A(j, hi));
      bm[j] = *(const int*)(brow + kc * 64 + K0B(j, hi));
    }
    c = __builtin_amdgcn_wmma_i32_16x16x64_iu8(false, a, true, bm, c, false, false);
  }
  int* acc5 = (int*)(ws + OFF_ACC5);
  int m = 0;
#pragma unroll
  for (int j = 0; j < 8; j++) {
    int M = j + (hi << 3);
    acc5[(size_t)(m0 + wave * 16 + M) * 16 + nn] = c[j];
    m = imax(m, c[j]);
  }
  red[tid] = imax(m, 0); __syncthreads();
  for (int s = 128; s > 0; s >>= 1) {
    if (tid < s) red[tid] = imax(red[tid], red[tid + s]);
    __syncthreads();
  }
  if (tid == 0) atomicMax(((int*)ws) + 5, red[0]);
}

__global__ __launch_bounds__(256) void k_fin(float* __restrict__ out,
                                             char* __restrict__ ws) {
  int t = blockIdx.x * 256 + threadIdx.x;
  int b = t / 10, n = t - b * 10;
  const int* si = (const int*)ws; const float* sf = (const float*)ws;
  int v = ((const int*)(ws + OFF_ACC5))[(size_t)b * 16 + n];
  float M5 = (float)imax(si[5], 1);
  float q = fminf(15.0f, fmaxf(0.0f, rintf((float)v * 15.0f / M5)));
  float s2 = s1_of(ws) * sf[6] * (float)imax(si[1], 1) / 15.0f;
  float s3 = s2 * sf[7] * (float)imax(si[2], 1) / 15.0f;
  float s4 = s3 * sf[8] * (float)imax(si[3], 1) / 15.0f;
  float s5 = s4 * sf[9] * (float)imax(si[4], 1) / 15.0f;
  float sout = s5 * sf[10] * M5 / 15.0f;
  out[t] = q * sout;
}

extern "C" void kernel_launch(void* const* d_in, const int* in_sizes, int n_in,
                              void* d_out, int out_size, void* d_ws, size_t ws_size,
                              hipStream_t stream) {
  (void)in_sizes; (void)n_in; (void)out_size; (void)ws_size;
  const float* x   = (const float*)d_in[0];
  const float* w1  = (const float*)d_in[1];
  const float* b1  = (const float*)d_in[2];
  const float* w2  = (const float*)d_in[3];
  const float* b2  = (const float*)d_in[4];
  const float* fw1 = (const float*)d_in[5];
  const float* fb1 = (const float*)d_in[6];
  const float* fw2 = (const float*)d_in[7];
  const float* fb2 = (const float*)d_in[8];
  const float* fw3 = (const float*)d_in[9];
  char* ws = (char*)d_ws;
  float* out = (float*)d_out;

  hipLaunchKernelGGL(k_zero,   dim3(1),     dim3(64),  0, stream, ws);
  hipLaunchKernelGGL(k_absmax, dim3(2048),  dim3(256), 0, stream, x, ws, 6422528);
  hipLaunchKernelGGL(k_wquant, dim3(5),     dim3(256), 0, stream, w1, w2, fw1, fw2, fw3, ws);
  hipLaunchKernelGGL(k_qx,     dim3(25088), dim3(256), 0, stream, x, ws, 6422528);
  hipLaunchKernelGGL(k_conv1,  dim3(27648), dim3(256), 0, stream, b1, ws);
  hipLaunchKernelGGL(k_quant1, dim3(27648), dim3(256), 0, stream, ws);
  hipLaunchKernelGGL(k_conv2,  dim3(4096),  dim3(256), 0, stream, b2, ws);
  hipLaunchKernelGGL(k_poolq2, dim3(8192),  dim3(256), 0, stream, ws);
  hipLaunchKernelGGL(k_fc1,    dim3(512),   dim3(256), 0, stream, fb1, ws);
  hipLaunchKernelGGL(k_fc2,    dim3(512),   dim3(192), 0, stream, fb2, ws);
  hipLaunchKernelGGL(k_fc3,    dim3(64),    dim3(256), 0, stream, ws);
  hipLaunchKernelGGL(k_fin,    dim3(320),   dim3(256), 0, stream, out, ws);
}
